// Mamba3_81406810128672
// MI455X (gfx1250) — matrix operand order repaired and checked by reading, not compiled, not run on hardware
//
#include <hip/hip_runtime.h>
#include <math.h>

// ---------------- problem constants ----------------
#define B_    2
#define L_    1024
#define DM_   1024
#define DIN_  2048
#define HD_   64
#define NH_   32
#define DS_   64
#define TH_   1024
#define DTOT  5312          // 2*DIN + 2*DS + NH + TH + NH
#define NTOK  (B_*L_)       // 2048
#define EPS_  1e-5f

// split offsets inside a zxbcdt row
#define OFF_Z   0
#define OFF_X   DIN_            // 2048
#define OFF_B   (2*DIN_)        // 4096
#define OFF_C   (2*DIN_+DS_)    // 4160
#define OFF_DT  (2*DIN_+2*DS_)  // 4224
#define OFF_TH  (OFF_DT+NH_)    // 4256
#define OFF_LAM (OFF_TH+TH_)    // 5280

typedef __attribute__((ext_vector_type(16))) __bf16 v16bf;
typedef __attribute__((ext_vector_type(8)))  __bf16 v8bf;
typedef __attribute__((ext_vector_type(8)))  float  v8f;
typedef int v4i_ __attribute__((__vector_size__(16)));

// gfx1250 async global->LDS copy (ASYNCcnt-tracked), with sync fallback
#if defined(__HIP_DEVICE_COMPILE__) && __has_builtin(__builtin_amdgcn_global_load_async_to_lds_b128)
#define HAVE_ASYNC_LDS 1
#endif

__device__ __forceinline__ void async_cp_b128(const __bf16* g, __bf16* l) {
#if defined(HAVE_ASYNC_LDS)
  __builtin_amdgcn_global_load_async_to_lds_b128(
      (__attribute__((address_space(1))) v4i_*)(uintptr_t)g,
      (__attribute__((address_space(3))) v4i_*)(unsigned int)(uintptr_t)l,
      0, 0);
#else
  *(v8bf*)l = *(const v8bf*)g;
#endif
}

__device__ __forceinline__ void wait_async_lds() {
#if defined(HAVE_ASYNC_LDS)
  asm volatile("s_wait_asynccnt 0" ::: "memory");
#endif
}

// ---------------- fp32 -> bf16 convert ----------------
__global__ void cvt_f32_bf16(const float* __restrict__ in, __bf16* __restrict__ out, int n) {
  int i = blockIdx.x * blockDim.x + threadIdx.x;
  if (i < n) out[i] = (__bf16)in[i];
}

// ---------------- bf16 WMMA GEMM: C[M,N] = A[M,K] @ W[N,K]^T ----------------
// block = 256 threads = 8 waves; block tile 128x64; BK=32; double-buffered LDS.
// wave tile 32x32: 2 A-frags x 2 B-frags -> 4 WMMAs per K-step.
// M multiple of 128, N multiple of 64, K multiple of 32.
#define LDA_ 40   // padded row stride (elements) to break LDS bank aliasing

__global__ __launch_bounds__(256)
void gemm_bf16_wmma(const __bf16* __restrict__ A, const __bf16* __restrict__ W,
                    float* __restrict__ C, int M, int N, int K) {
  __shared__ __align__(16) __bf16 As[2][128 * LDA_];
  __shared__ __align__(16) __bf16 Bs[2][64 * LDA_];

  const int tid  = threadIdx.x;
  const int lane = tid & 31;
  const int wave = tid >> 5;
  const int hi   = (lane >= 16) ? 1 : 0;   // lane half selects K sub-range per ISA layout
  const int l16  = lane & 15;

  const int bm = blockIdx.y * 128;
  const int bn = blockIdx.x * 64;
  const int wm = (wave & 3) * 32;          // wave row group within block tile
  const int wn = (wave >> 2) * 32;         // wave col group within block tile

  // cooperative staging map: A = 512 b128 chunks, B = 256 b128 chunks
  const int arow0 = tid >> 1;                    // chunks 2tid,2tid+1 and 2tid+256,2tid+257
  const int akc0  = (tid & 1) * 2;               // A: 2 chunks per (thread, half)
  const int brow  = tid >> 2;
  const int bkc   = tid & 3;

  const __bf16* __restrict__ ag0 = A + (size_t)(bm + arow0) * K;        // rows tid>>1
  const __bf16* __restrict__ ag1 = A + (size_t)(bm + 64 + arow0) * K;   // rows +64
  const __bf16* __restrict__ bg  = W + (size_t)(bn + brow) * K;

  v8f c00 = {}; v8f c01 = {}; v8f c10 = {}; v8f c11 = {};

  const int nk = K >> 5;

  // prologue: stage K-step 0 into buffer 0
  {
    async_cp_b128(ag0 + akc0 * 8,       &As[0][arow0 * LDA_ + akc0 * 8]);
    async_cp_b128(ag0 + akc0 * 8 + 8,   &As[0][arow0 * LDA_ + akc0 * 8 + 8]);
    async_cp_b128(ag1 + akc0 * 8,       &As[0][(64 + arow0) * LDA_ + akc0 * 8]);
    async_cp_b128(ag1 + akc0 * 8 + 8,   &As[0][(64 + arow0) * LDA_ + akc0 * 8 + 8]);
    async_cp_b128(bg + bkc * 8,         &Bs[0][brow * LDA_ + bkc * 8]);
  }

  for (int kt = 0; kt < nk; ++kt) {
    const int buf = kt & 1;
    wait_async_lds();
    __syncthreads();                       // staged tile visible to all waves

    // stage next K-step into the other buffer (overlaps with WMMA below)
    if (kt + 1 < nk) {
      const int kk = (kt + 1) << 5;
      const int nb = buf ^ 1;
      async_cp_b128(ag0 + kk + akc0 * 8,     &As[nb][arow0 * LDA_ + akc0 * 8]);
      async_cp_b128(ag0 + kk + akc0 * 8 + 8, &As[nb][arow0 * LDA_ + akc0 * 8 + 8]);
      async_cp_b128(ag1 + kk + akc0 * 8,     &As[nb][(64 + arow0) * LDA_ + akc0 * 8]);
      async_cp_b128(ag1 + kk + akc0 * 8 + 8, &As[nb][(64 + arow0) * LDA_ + akc0 * 8 + 8]);
      async_cp_b128(bg + kk + bkc * 8,       &Bs[nb][brow * LDA_ + bkc * 8]);
      if (kt + 2 < nk) {                   // pull K-step+2 toward near caches
        __builtin_prefetch(ag0 + kk + 32, 0, 3);
        __builtin_prefetch(ag1 + kk + 32, 0, 3);
        __builtin_prefetch(bg + kk + 32, 0, 3);
      }
    }

    // fragments from LDS (ds_load_b128 pairs), per documented 16-bit layouts
    const __bf16* ar0 = &As[buf][(wm + l16) * LDA_];
    const __bf16* ar1 = &As[buf][(wm + 16 + l16) * LDA_];
    const __bf16* br0 = &Bs[buf][(wn + l16) * LDA_];
    const __bf16* br1 = &Bs[buf][(wn + 16 + l16) * LDA_];

    v8bf t0, t1;
    t0 = *(const v8bf*)(ar0 + hi * 8); t1 = *(const v8bf*)(ar0 + 16 + hi * 8);
    v16bf a0 = __builtin_shufflevector(t0, t1, 0,1,2,3,4,5,6,7,8,9,10,11,12,13,14,15);
    t0 = *(const v8bf*)(ar1 + hi * 8); t1 = *(const v8bf*)(ar1 + 16 + hi * 8);
    v16bf a1 = __builtin_shufflevector(t0, t1, 0,1,2,3,4,5,6,7,8,9,10,11,12,13,14,15);
    t0 = *(const v8bf*)(br0 + hi * 8); t1 = *(const v8bf*)(br0 + 16 + hi * 8);
    v16bf b0 = __builtin_shufflevector(t0, t1, 0,1,2,3,4,5,6,7,8,9,10,11,12,13,14,15);
    t0 = *(const v8bf*)(br1 + hi * 8); t1 = *(const v8bf*)(br1 + 16 + hi * 8);
    v16bf b1 = __builtin_shufflevector(t0, t1, 0,1,2,3,4,5,6,7,8,9,10,11,12,13,14,15);

    c00 = __builtin_amdgcn_wmma_f32_16x16x32_bf16(false, a0, false, b0, (short)0, c00, false, false);
    c01 = __builtin_amdgcn_wmma_f32_16x16x32_bf16(false, a0, false, b1, (short)0, c01, false, false);
    c10 = __builtin_amdgcn_wmma_f32_16x16x32_bf16(false, a1, false, b0, (short)0, c10, false, false);
    c11 = __builtin_amdgcn_wmma_f32_16x16x32_bf16(false, a1, false, b1, (short)0, c11, false, false);

    __syncthreads();                       // LDS reads done before buffer reuse
  }

  // C/D layout: VGPR i holds row (tile_m + i + hi*8), column l16
  const int cm0 = bm + wm;
  const int cn0 = bn + wn;
  #pragma unroll
  for (int i = 0; i < 8; ++i) {
    const size_t r0 = (size_t)(cm0 + hi * 8 + i) * N;
    const size_t r1 = (size_t)(cm0 + 16 + hi * 8 + i) * N;
    C[r0 + cn0 + l16]      = c00[i];
    C[r0 + cn0 + 16 + l16] = c01[i];
    C[r1 + cn0 + l16]      = c10[i];
    C[r1 + cn0 + 16 + l16] = c11[i];
  }
}

// ---------------- per-token prep: dt/alpha/beta/gamma + RMS(B), RMS(C) ----------------
__global__ __launch_bounds__(64)
void token_prep(const float* __restrict__ zx, const float* __restrict__ dt_bias,
                const float* __restrict__ A_log, const float* __restrict__ Bnw,
                const float* __restrict__ Cnw,
                float* __restrict__ alpha, float* __restrict__ beta,
                float* __restrict__ gamma, float* __restrict__ Bg, float* __restrict__ Cg) {
  const int tok = blockIdx.x;
  const int tid = threadIdx.x;            // 0..63
  const float* row = zx + (size_t)tok * DTOT;
  __shared__ float rb[64], rc[64], ms[2];

  float bv = row[OFF_B + tid];
  float cv = row[OFF_C + tid];
  rb[tid] = bv * bv;
  rc[tid] = cv * cv;
  __syncthreads();
  if (tid == 0) {
    float s0 = 0.f, s1 = 0.f;
    for (int i = 0; i < 64; ++i) { s0 += rb[i]; s1 += rc[i]; }
    ms[0] = s0 * (1.0f / 64.0f);
    ms[1] = s1 * (1.0f / 64.0f);
  }
  __syncthreads();
  Bg[(size_t)tok * DS_ + tid] = bv * rsqrtf(ms[0] + EPS_) * Bnw[tid];
  Cg[(size_t)tok * DS_ + tid] = cv * rsqrtf(ms[1] + EPS_) * Cnw[tid];

  if (tid < NH_) {
    const int h = tid;
    float d  = row[OFF_DT + h] + dt_bias[h];
    float dt = (d > 20.f) ? d : log1pf(expf(d));         // softplus
    float Ax = -expf(A_log[h]);
    float al = expf(dt * Ax);
    float lam = 1.f / (1.f + expf(-row[OFF_LAM + h]));   // sigmoid
    alpha[(size_t)tok * NH_ + h] = al;
    gamma[(size_t)tok * NH_ + h] = lam * dt;
    beta [(size_t)tok * NH_ + h] = (1.f - lam) * dt * al;
  }
}

// ---------------- sequential SSM scan: one workgroup per (b,h) ----------------
// h-state 64x64 + bx_prev in registers: thread owns (p = tid>>2, n in [(tid&3)*16, +16))
__global__ __launch_bounds__(256)
void ssm_scan(const float* __restrict__ zx, const float* __restrict__ alpha,
              const float* __restrict__ beta, const float* __restrict__ gamma,
              const float* __restrict__ Bg, const float* __restrict__ Cg,
              const float* __restrict__ B_bias, const float* __restrict__ C_bias,
              const float* __restrict__ Dvec, float* __restrict__ y) {
  const int b = blockIdx.x / NH_;
  const int h = blockIdx.x % NH_;
  const int tid = threadIdx.x;
  const int p  = tid >> 2;
  const int nq = tid & 3;

  __shared__ float xs[HD_], Bpre[DS_], Cpre[DS_], Bh[DS_], Ch[DS_];
  __shared__ float thcs[DS_ / 2], cA[DS_ / 2], sA[DS_ / 2];
  __shared__ float sc[3];

  if (tid < DS_ / 2) thcs[tid] = 0.f;
  const float Dh = Dvec[h];

  float hst[16], bxp[16];
  #pragma unroll
  for (int j = 0; j < 16; ++j) { hst[j] = 0.f; bxp[j] = 0.f; }

  for (int t = 0; t < L_; ++t) {
    const size_t tok = (size_t)b * L_ + t;
    const float* row = zx + tok * DTOT;
    __syncthreads();                              // prev-iter LDS reads done
    if (tid < 64) {
      xs[tid] = row[OFF_X + h * HD_ + tid];
    } else if (tid < 128) {
      int i = tid - 64;
      Bpre[i] = Bg[tok * DS_ + i] + B_bias[h * DS_ + i];
    } else if (tid < 192) {
      int i = tid - 128;
      Cpre[i] = Cg[tok * DS_ + i] + C_bias[h * DS_ + i];
    } else if (tid < 224) {
      int i = tid - 192;
      float v = thcs[i] + row[OFF_TH + h * (DS_ / 2) + i];   // theta cumsum
      thcs[i] = v;
      cA[i] = cosf(v);
      sA[i] = sinf(v);
    } else if (tid == 224) {
      sc[0] = alpha[tok * NH_ + h];
      sc[1] = beta [tok * NH_ + h];
      sc[2] = gamma[tok * NH_ + h];
    }
    __syncthreads();
    if (tid < 32) {                               // rotate B
      float v1 = Bpre[tid], v2 = Bpre[tid + 32];
      Bh[tid]      = v1 * cA[tid] - v2 * sA[tid];
      Bh[tid + 32] = v1 * sA[tid] + v2 * cA[tid];
    } else if (tid < 64) {                        // rotate C
      int i = tid - 32;
      float v1 = Cpre[i], v2 = Cpre[i + 32];
      Ch[i]      = v1 * cA[i] - v2 * sA[i];
      Ch[i + 32] = v1 * sA[i] + v2 * cA[i];
    }
    __syncthreads();
    const float a = sc[0], bcf = sc[1], g = sc[2];
    const float xp = xs[p];
    float part = 0.f;
    #pragma unroll
    for (int j = 0; j < 16; ++j) {
      const int n = nq * 16 + j;
      float bx = xp * Bh[n];
      float hv = a * hst[j] + bcf * bxp[j] + g * bx;
      hst[j] = hv;
      bxp[j] = bx;
      part += Ch[n] * hv;
    }
    part += __shfl_xor(part, 1, 32);              // reduce over the 4 n-slices
    part += __shfl_xor(part, 2, 32);
    if (nq == 0) y[tok * DIN_ + h * HD_ + p] = part + Dh * xp;
  }
}

// ---------------- gate (silu) + RMS norm, emit bf16 ----------------
__global__ __launch_bounds__(256)
void gate_norm(const float* __restrict__ y, const float* __restrict__ zx,
               const float* __restrict__ norm_w, __bf16* __restrict__ gout) {
  const int tok = blockIdx.x;
  const int tid = threadIdx.x;
  const float* yr = y + (size_t)tok * DIN_;
  const float* zr = zx + (size_t)tok * DTOT;      // z is split 0
  __shared__ float red[256];
  __shared__ float inv;

  float gv[8];
  float ss = 0.f;
  #pragma unroll
  for (int k = 0; k < 8; ++k) {
    int i = tid + k * 256;
    float z = zr[OFF_Z + i];
    float g = yr[i] * z * (1.f / (1.f + expf(-z)));  // y * silu(z)
    gv[k] = g;
    ss += g * g;
  }
  red[tid] = ss;
  __syncthreads();
  for (int s = 128; s > 0; s >>= 1) {
    if (tid < s) red[tid] += red[tid + s];
    __syncthreads();
  }
  if (tid == 0) inv = rsqrtf(red[0] * (1.0f / (float)DIN_) + EPS_);
  __syncthreads();
  #pragma unroll
  for (int k = 0; k < 8; ++k) {
    int i = tid + k * 256;
    gout[(size_t)tok * DIN_ + i] = (__bf16)(gv[k] * inv * norm_w[i]);
  }
}

// ---------------- host side ----------------
extern "C" void kernel_launch(void* const* d_in, const int* in_sizes, int n_in,
                              void* d_out, int out_size, void* d_ws, size_t ws_size,
                              hipStream_t stream) {
  const float* u       = (const float*)d_in[0];
  const float* in_w    = (const float*)d_in[1];
  const float* dt_bias = (const float*)d_in[2];
  const float* A_log   = (const float*)d_in[3];
  const float* Dv      = (const float*)d_in[4];
  const float* Bnw     = (const float*)d_in[5];
  const float* Cnw     = (const float*)d_in[6];
  const float* Bb      = (const float*)d_in[7];
  const float* Cb      = (const float*)d_in[8];
  const float* nw      = (const float*)d_in[9];
  const float* out_w   = (const float*)d_in[10];
  (void)in_sizes; (void)n_in; (void)out_size; (void)ws_size;

  char* ws = (char*)d_ws;
  size_t off = 0;
  auto carve = [&](size_t bytes) -> void* {
    void* p = ws + off;
    off = (off + bytes + 255) & ~(size_t)255;
    return p;
  };

  __bf16* u_bf   = (__bf16*)carve((size_t)NTOK * DM_ * 2);
  __bf16* wi_bf  = (__bf16*)carve((size_t)DTOT * DM_ * 2);
  __bf16* wo_bf  = (__bf16*)carve((size_t)DM_ * DIN_ * 2);
  float*  zx     = (float*) carve((size_t)NTOK * DTOT * 4);
  float*  alpha  = (float*) carve((size_t)NTOK * NH_ * 4);
  float*  beta   = (float*) carve((size_t)NTOK * NH_ * 4);
  float*  gamma  = (float*) carve((size_t)NTOK * NH_ * 4);
  float*  Bg     = (float*) carve((size_t)NTOK * DS_ * 4);
  float*  Cg     = (float*) carve((size_t)NTOK * DS_ * 4);
  float*  yb     = (float*) carve((size_t)NTOK * DIN_ * 4);
  __bf16* g_bf   = (__bf16*)carve((size_t)NTOK * DIN_ * 2);

  // 1) bf16 conversions
  {
    int n0 = NTOK * DM_;
    cvt_f32_bf16<<<(n0 + 255) / 256, 256, 0, stream>>>(u, u_bf, n0);
    int n1 = DTOT * DM_;
    cvt_f32_bf16<<<(n1 + 255) / 256, 256, 0, stream>>>(in_w, wi_bf, n1);
    int n2 = DM_ * DIN_;
    cvt_f32_bf16<<<(n2 + 255) / 256, 256, 0, stream>>>(out_w, wo_bf, n2);
  }

  // 2) zxbcdt = u @ in_proj_w^T   (M=2048, N=5312, K=1024)
  gemm_bf16_wmma<<<dim3(DTOT / 64, NTOK / 128), 256, 0, stream>>>(
      u_bf, wi_bf, zx, NTOK, DTOT, DM_);

  // 3) per-token scalars + group RMS norms
  token_prep<<<NTOK, 64, 0, stream>>>(zx, dt_bias, A_log, Bnw, Cnw,
                                      alpha, beta, gamma, Bg, Cg);

  // 4) sequential scan, one WG per (b, h)
  ssm_scan<<<B_ * NH_, 256, 0, stream>>>(zx, alpha, beta, gamma, Bg, Cg,
                                         Bb, Cb, Dv, yb);

  // 5) gate + final RMS norm -> bf16
  gate_norm<<<NTOK, 256, 0, stream>>>(yb, zx, nw, g_bf);

  // 6) out = g @ out_proj_w^T   (M=2048, N=1024, K=2048)
  gemm_bf16_wmma<<<dim3(DM_ / 64, NTOK / 128), 256, 0, stream>>>(
      g_bf, wo_bf, (float*)d_out, NTOK, DM_, DIN_);
}
